// MapNetBehavior_5738076307532
// MI455X (gfx1250) — matrix operand order repaired and checked by reading, not compiled, not run on hardware
//
#include <hip/hip_runtime.h>
#include <hip/hip_bf16.h>

// ---------------- problem constants (from reference) ----------------
#define N_NODES 150000
#define DCH     128
#define NSCALE  6
#define NEDGE   150000
#define NEL     20000
#define NLAYER  4
#define GN_EPS  1e-5f
#define WPB     4       // waves per block in GEMM kernels

typedef __bf16 bf16;
typedef __attribute__((ext_vector_type(4)))  __bf16 v4bf;
typedef __attribute__((ext_vector_type(8)))  __bf16 v8bf;
typedef __attribute__((ext_vector_type(16))) __bf16 v16bf;
typedef __attribute__((ext_vector_type(8)))  float  v8f;

// ---------------- WMMA fragment loaders (CDNA5 wave32 layouts) ----------------
// A (16x32 bf16, MxK): lane m=lane&15, hi=lane>>4.
//   halves 0..15 = K in [hi*16, hi*16+16) (contiguous, matching B)
__device__ __forceinline__ v16bf load_a_frag(const bf16* __restrict__ rowptr, int kb, int hi) {
  v8bf lo = *reinterpret_cast<const v8bf*>(rowptr + kb + hi * 16);
  v8bf hv = *reinterpret_cast<const v8bf*>(rowptr + kb + hi * 16 + 8);
  return __builtin_shufflevector(lo, hv, 0,1,2,3,4,5,6,7,8,9,10,11,12,13,14,15);
}

// B (32x16 bf16, KxN): lane holds column n=lane&15, 16 consecutive K from (lane>>4)*16.
// Weight staged in LDS pre-transposed as wT[n][k] -> two contiguous ds_load_b128.
__device__ __forceinline__ v16bf lds_b_frag(const bf16* wlds, int ncol, int kb, int hi) {
  const bf16* base = wlds + ncol * DCH + kb + hi * 16;
  v8bf lo = *reinterpret_cast<const v8bf*>(base);
  v8bf hv = *reinterpret_cast<const v8bf*>(base + 8);
  return __builtin_shufflevector(lo, hv, 0,1,2,3,4,5,6,7,8,9,10,11,12,13,14,15);
}

// Async-stage one 128x128 bf16 weight (32KB) into LDS: 2048 16B chunks over 128 threads.
// Uses CDNA5 global_load_async_to_lds_b128 (ASYNCcnt), no VGPR round-trip.
__device__ __forceinline__ void stage_weight(const bf16* __restrict__ wT, bf16* wlds) {
  unsigned ldsbase = (unsigned)(uintptr_t)(&wlds[0]);
  const char* g = (const char*)wT;
#pragma unroll
  for (int i = 0; i < 16; ++i) {
    int chunk = (int)threadIdx.x + i * 128;
    unsigned loff = ldsbase + chunk * 16;
    const char* gaddr = g + chunk * 16;
    asm volatile("global_load_async_to_lds_b128 %0, %1, off"
                 :: "v"(loff), "v"(gaddr) : "memory");
  }
  asm volatile("s_wait_asynccnt 0x0" ::: "memory");
  __syncthreads();
}

// 16-row x 128-col tile: A register-resident, B double-buffered from LDS, 32 WMMAs.
__device__ __forceinline__ void tile_mma(const bf16* __restrict__ arow, const bf16* wlds,
                                         int m, int hi, v8f (&acc)[8]) {
  v16bf a[4];
#pragma unroll
  for (int ks = 0; ks < 4; ++ks) a[ks] = load_a_frag(arow, ks * 32, hi);
  v16bf bcur = lds_b_frag(wlds, m, 0, hi);  // (t=0, ks=0)
#pragma unroll
  for (int i = 0; i < 32; ++i) {
    int t = i >> 2, ks = i & 3;
    int j = (i + 1) & 31;
    v16bf bnext = lds_b_frag(wlds, (j >> 2) * 16 + m, (j & 3) * 32, hi);
    acc[t] = __builtin_amdgcn_wmma_f32_16x16x32_bf16(false, a[ks], false, bcur,
                                                     (short)0, acc[t], false, false);
    bcur = bnext;
  }
}

// ---------------- node GEMM: out[r][:] = A[r][:] @ W  (persistent, LDS-staged W) ----------------
__global__ void __launch_bounds__(128) gemm_node(const bf16* __restrict__ abf,
                                                 const bf16* __restrict__ wT,
                                                 float* __restrict__ out, int ntiles) {
  __shared__ alignas(16) bf16 wlds[DCH * DCH];
  stage_weight(wT, wlds);
  int lane = threadIdx.x & 31;
  int m = lane & 15, hi = lane >> 4;
  int totalWaves = gridDim.x * WPB;
  for (int tile = blockIdx.x * WPB + (threadIdx.x >> 5); tile < ntiles; tile += totalWaves) {
    int r0 = tile * 16;
    const bf16* arow = abf + (size_t)(r0 + m) * DCH;
    v8f acc[8] = {};
    tile_mma(arow, wlds, m, hi, acc);
    // D layout: VGPR r -> row (r + 8*hi), col (t*16 + m)
#pragma unroll
    for (int t = 0; t < 8; ++t)
#pragma unroll
      for (int r = 0; r < 8; ++r)
        out[(size_t)(r0 + r + 8 * hi) * DCH + t * 16 + m] = acc[t][r];
  }
}

// ---------------- edge GEMM: temp[u[e]][:] += feat[v[e]][:] @ W ----------------
__global__ void __launch_bounds__(128) gemm_edge(const bf16* __restrict__ featbf,
                                                 const bf16* __restrict__ wT,
                                                 const int* __restrict__ eu,
                                                 const int* __restrict__ ev,
                                                 float* __restrict__ temp, int ntiles) {
  __shared__ alignas(16) bf16 wlds[DCH * DCH];
  stage_weight(wT, wlds);
  int lane = threadIdx.x & 31;
  int m = lane & 15, hi = lane >> 4;
  int totalWaves = gridDim.x * WPB;
  for (int tile = blockIdx.x * WPB + (threadIdx.x >> 5); tile < ntiles; tile += totalWaves) {
    int e0 = tile * 16;
    int vsrc = ev[e0 + m];
    int uidx[8];
#pragma unroll
    for (int r = 0; r < 8; ++r) uidx[r] = eu[e0 + r + 8 * hi];  // prefetch scatter targets
    const bf16* arow = featbf + (size_t)vsrc * DCH;
    v8f acc[8] = {};
    tile_mma(arow, wlds, m, hi, acc);
#pragma unroll
    for (int t = 0; t < 8; ++t)
#pragma unroll
      for (int r = 0; r < 8; ++r)
        unsafeAtomicAdd(&temp[(size_t)uidx[r] * DCH + t * 16 + m], acc[t][r]);
  }
}

// ---------------- wave32 reductions ----------------
__device__ __forceinline__ float wave_sum(float v) {
#pragma unroll
  for (int off = 16; off > 0; off >>= 1) v += __shfl_xor(v, off, 32);
  return v;
}

__device__ __forceinline__ void store4bf(bf16* __restrict__ base, float a, float b,
                                         float c, float d) {
  v4bf o;
  o[0] = (__bf16)a; o[1] = (__bf16)b; o[2] = (__bf16)c; o[3] = (__bf16)d;
  *reinterpret_cast<v4bf*>(base) = o;
}

// ---------------- stem: h = relu(in(Nx2) @ w1(2x128) + b1) -> bf16 ----------------
__global__ void __launch_bounds__(256) stem_lin1(const float* __restrict__ in,
                                                 const float* __restrict__ w1,
                                                 const float* __restrict__ b1,
                                                 bf16* __restrict__ out, int total) {
  int id = blockIdx.x * 256 + threadIdx.x;
  if (id >= total) return;
  int n = id >> 7, d = id & 127;
  float v = fmaf(in[2 * n], w1[d], fmaf(in[2 * n + 1], w1[DCH + d], b1[d]));
  out[id] = (__bf16)fmaxf(v, 0.f);
}

// ---------------- stem combine: feat = relu(GN(x) + GN(y)) ----------------
__global__ void __launch_bounds__(256) stem_combine(const float* __restrict__ x,
                                                    const float* __restrict__ y,
                                                    const float* __restrict__ gwx,
                                                    const float* __restrict__ gbx,
                                                    const float* __restrict__ gwy,
                                                    const float* __restrict__ gby,
                                                    float* __restrict__ feat,
                                                    bf16* __restrict__ featbf, int rows) {
  int row = blockIdx.x * 8 + (threadIdx.x >> 5);
  if (row >= rows) return;
  int lane = threadIdx.x & 31;
  const float4 xv = reinterpret_cast<const float4*>(x + (size_t)row * DCH)[lane];
  const float4 yv = reinterpret_cast<const float4*>(y + (size_t)row * DCH)[lane];
  float sx = xv.x + xv.y + xv.z + xv.w;
  float sxx = xv.x * xv.x + xv.y * xv.y + xv.z * xv.z + xv.w * xv.w;
  float sy = yv.x + yv.y + yv.z + yv.w;
  float syy = yv.x * yv.x + yv.y * yv.y + yv.z * yv.z + yv.w * yv.w;
  sx = wave_sum(sx); sxx = wave_sum(sxx); sy = wave_sum(sy); syy = wave_sum(syy);
  float mx = sx * (1.f / DCH), my = sy * (1.f / DCH);
  float ivx = rsqrtf(sxx * (1.f / DCH) - mx * mx + GN_EPS);
  float ivy = rsqrtf(syy * (1.f / DCH) - my * my + GN_EPS);
  int c = lane * 4;
  float f[4];
  const float xs[4] = {xv.x, xv.y, xv.z, xv.w};
  const float ys[4] = {yv.x, yv.y, yv.z, yv.w};
#pragma unroll
  for (int i = 0; i < 4; ++i) {
    float gx = (xs[i] - mx) * ivx * gwx[c + i] + gbx[c + i];
    float gy = (ys[i] - my) * ivy * gwy[c + i] + gby[c + i];
    f[i] = fmaxf(gx + gy, 0.f);
  }
  float4* fo = reinterpret_cast<float4*>(feat + (size_t)row * DCH) + lane;
  *fo = make_float4(f[0], f[1], f[2], f[3]);
  store4bf(featbf + (size_t)row * DCH + c, f[0], f[1], f[2], f[3]);
}

// ---------------- mid-layer: featbf = bf16(relu(GN(temp))) ----------------
__global__ void __launch_bounds__(256) gn_relu_bf(const float* __restrict__ x,
                                                  const float* __restrict__ gw,
                                                  const float* __restrict__ gb,
                                                  bf16* __restrict__ outbf, int rows) {
  int row = blockIdx.x * 8 + (threadIdx.x >> 5);
  if (row >= rows) return;
  int lane = threadIdx.x & 31;
  const float4 xv = reinterpret_cast<const float4*>(x + (size_t)row * DCH)[lane];
  float s = xv.x + xv.y + xv.z + xv.w;
  float s2 = xv.x * xv.x + xv.y * xv.y + xv.z * xv.z + xv.w * xv.w;
  s = wave_sum(s); s2 = wave_sum(s2);
  float mean = s * (1.f / DCH);
  float inv = rsqrtf(s2 * (1.f / DCH) - mean * mean + GN_EPS);
  int c = lane * 4;
  const float xs[4] = {xv.x, xv.y, xv.z, xv.w};
  float f[4];
#pragma unroll
  for (int i = 0; i < 4; ++i)
    f[i] = fmaxf((xs[i] - mean) * inv * gw[c + i] + gb[c + i], 0.f);
  store4bf(outbf + (size_t)row * DCH + c, f[0], f[1], f[2], f[3]);
}

// ---------------- end-of-layer: feat = relu(GN(t2) + res); featbf mirror ----------------
__global__ void __launch_bounds__(256) gn_add(const float* __restrict__ x,
                                              const float* __restrict__ gw,
                                              const float* __restrict__ gb,
                                              float* __restrict__ feat,  // res in, feat out
                                              bf16* __restrict__ featbf,
                                              float* __restrict__ out_final, int rows) {
  int row = blockIdx.x * 8 + (threadIdx.x >> 5);
  if (row >= rows) return;
  int lane = threadIdx.x & 31;
  const float4 xv = reinterpret_cast<const float4*>(x + (size_t)row * DCH)[lane];
  float s = xv.x + xv.y + xv.z + xv.w;
  float s2 = xv.x * xv.x + xv.y * xv.y + xv.z * xv.z + xv.w * xv.w;
  s = wave_sum(s); s2 = wave_sum(s2);
  float mean = s * (1.f / DCH);
  float inv = rsqrtf(s2 * (1.f / DCH) - mean * mean + GN_EPS);
  int c = lane * 4;
  float4* fp = reinterpret_cast<float4*>(feat + (size_t)row * DCH) + lane;
  float4 rv = *fp;  // residual (== feat at top of layer)
  const float xs[4] = {xv.x, xv.y, xv.z, xv.w};
  const float rs[4] = {rv.x, rv.y, rv.z, rv.w};
  float f[4];
#pragma unroll
  for (int i = 0; i < 4; ++i)
    f[i] = fmaxf((xs[i] - mean) * inv * gw[c + i] + gb[c + i] + rs[i], 0.f);
  *fp = make_float4(f[0], f[1], f[2], f[3]);
  store4bf(featbf + (size_t)row * DCH + c, f[0], f[1], f[2], f[3]);
  if (out_final) {
    float4* op = reinterpret_cast<float4*>(out_final + (size_t)row * DCH) + lane;
    *op = make_float4(f[0], f[1], f[2], f[3]);
  }
}

// ---------------- weight convert+transpose: wT[m][n*128+k] = bf16(w[m][k*128+n]) ----------------
__global__ void __launch_bounds__(256) conv_wT(const float* __restrict__ src,
                                               bf16* __restrict__ dst, int total) {
  int id = blockIdx.x * 256 + threadIdx.x;
  if (id >= total) return;
  int mat = id >> 14, rem = id & 16383;
  int n = rem >> 7, k = rem & 127;
  dst[id] = (__bf16)src[(mat << 14) + (k << 7) + n];
}

// ---------------- host orchestration ----------------
extern "C" void kernel_launch(void* const* d_in, const int* in_sizes, int n_in,
                              void* d_out, int out_size, void* d_ws, size_t ws_size,
                              hipStream_t stream) {
  const float* ctrs      = (const float*)d_in[0];
  const float* feats_in  = (const float*)d_in[1];
  const float* w_in1     = (const float*)d_in[2];
  const float* b_in1     = (const float*)d_in[3];
  const float* w_in2     = (const float*)d_in[4];
  const float* gn_in_w   = (const float*)d_in[5];
  const float* gn_in_b   = (const float*)d_in[6];
  const float* w_seg1    = (const float*)d_in[7];
  const float* b_seg1    = (const float*)d_in[8];
  const float* w_seg2    = (const float*)d_in[9];
  const float* gn_seg_w  = (const float*)d_in[10];
  const float* gn_seg_b  = (const float*)d_in[11];
  const float* w_ctr     = (const float*)d_in[12];
  const float* w_pre     = (const float*)d_in[13];
  const float* w_suc     = (const float*)d_in[14];
  const float* w_left    = (const float*)d_in[15];
  const float* w_right   = (const float*)d_in[16];
  const float* gn_norm_w = (const float*)d_in[17];
  const float* gn_norm_b = (const float*)d_in[18];
  const float* w_ctr2    = (const float*)d_in[19];
  const float* gn_ctr2_w = (const float*)d_in[20];
  const float* gn_ctr2_b = (const float*)d_in[21];
  const int* pre_u  = (const int*)d_in[22];
  const int* pre_v  = (const int*)d_in[23];
  const int* suc_u  = (const int*)d_in[24];
  const int* suc_v  = (const int*)d_in[25];
  const int* left_u = (const int*)d_in[26];
  const int* left_v = (const int*)d_in[27];
  const int* right_u = (const int*)d_in[28];
  const int* right_v = (const int*)d_in[29];
  (void)in_sizes; (void)n_in; (void)out_size; (void)ws_size;

  const size_t ND4 = (size_t)N_NODES * DCH * sizeof(float);
  const size_t ND2 = (size_t)N_NODES * DCH * sizeof(bf16);
  char* ws = (char*)d_ws;
  float* featF = (float*)ws; ws += ND4;   // feat (== res at layer top)
  float* tempF = (float*)ws; ws += ND4;   // aggregation buffer
  bf16*  featB = (bf16*)ws;  ws += ND2;   // bf16 mirror of current GEMM input
  bf16*  auxB  = (bf16*)ws;  ws += ND2;   // stem scratch
  bf16*  wB    = (bf16*)ws;               // 66 transposed bf16 weight mats
  float* t2F   = (float*)d_out;           // second-GEMM scratch; final layer leaves result

  const int MSZ = DCH * DCH;  // 16384
  bf16* wb_in2   = wB + 0  * MSZ;
  bf16* wb_seg2  = wB + 1  * MSZ;
  bf16* wb_ctr   = wB + 2  * MSZ;   // 4
  bf16* wb_pre   = wB + 6  * MSZ;   // 24
  bf16* wb_suc   = wB + 30 * MSZ;   // 24
  bf16* wb_left  = wB + 54 * MSZ;   // 4
  bf16* wb_right = wB + 58 * MSZ;   // 4
  bf16* wb_ctr2  = wB + 62 * MSZ;   // 4

  auto cvt = [&](const float* s, bf16* d, int nm) {
    int tot = nm * MSZ;
    conv_wT<<<(tot + 255) / 256, 256, 0, stream>>>(s, d, tot);
  };
  cvt(w_in2, wb_in2, 1);   cvt(w_seg2, wb_seg2, 1);
  cvt(w_ctr, wb_ctr, 4);   cvt(w_pre, wb_pre, 24); cvt(w_suc, wb_suc, 24);
  cvt(w_left, wb_left, 4); cvt(w_right, wb_right, 4); cvt(w_ctr2, wb_ctr2, 4);

  const int totND = N_NODES * DCH;
  const int nodeTiles = N_NODES / 16;   // 9375
  const int edgeTiles = NEDGE / 16;     // 9375
  const int elTiles   = NEL / 16;       // 1250
  auto gblocks = [](int ntiles) {
    int b = (ntiles + WPB - 1) / WPB;
    return b > 1024 ? 1024 : b;         // persistent waves amortize LDS staging
  };
  const int nodeB = gblocks(nodeTiles), edgeB = gblocks(edgeTiles), elB = gblocks(elTiles);
  const int gnBlocks = (N_NODES + 7) / 8;

  // ---- stems ----
  stem_lin1<<<(totND + 255) / 256, 256, 0, stream>>>(ctrs, w_in1, b_in1, featB, totND);
  stem_lin1<<<(totND + 255) / 256, 256, 0, stream>>>(feats_in, w_seg1, b_seg1, auxB, totND);
  gemm_node<<<nodeB, 128, 0, stream>>>(featB, wb_in2, t2F, nodeTiles);   // x
  gemm_node<<<nodeB, 128, 0, stream>>>(auxB, wb_seg2, tempF, nodeTiles); // y
  stem_combine<<<gnBlocks, 256, 0, stream>>>(t2F, tempF, gn_in_w, gn_in_b,
                                             gn_seg_w, gn_seg_b, featF, featB, N_NODES);

  // ---- fuse layers ----
  for (int l = 0; l < NLAYER; ++l) {
    gemm_node<<<nodeB, 128, 0, stream>>>(featB, wb_ctr + l * MSZ, tempF, nodeTiles);
    for (int s = 0; s < NSCALE; ++s) {
      gemm_edge<<<edgeB, 128, 0, stream>>>(featB, wb_pre + (l * NSCALE + s) * MSZ,
                                           pre_u + (size_t)s * NEDGE,
                                           pre_v + (size_t)s * NEDGE, tempF, edgeTiles);
      gemm_edge<<<edgeB, 128, 0, stream>>>(featB, wb_suc + (l * NSCALE + s) * MSZ,
                                           suc_u + (size_t)s * NEDGE,
                                           suc_v + (size_t)s * NEDGE, tempF, edgeTiles);
    }
    gemm_edge<<<elB, 128, 0, stream>>>(featB, wb_left + l * MSZ, left_u, left_v, tempF, elTiles);
    gemm_edge<<<elB, 128, 0, stream>>>(featB, wb_right + l * MSZ, right_u, right_v, tempF, elTiles);
    gn_relu_bf<<<gnBlocks, 256, 0, stream>>>(tempF, gn_norm_w + l * DCH, gn_norm_b + l * DCH,
                                             featB, N_NODES);
    gemm_node<<<nodeB, 128, 0, stream>>>(featB, wb_ctr2 + l * MSZ, t2F, nodeTiles);
    gn_add<<<gnBlocks, 256, 0, stream>>>(t2F, gn_ctr2_w + l * DCH, gn_ctr2_b + l * DCH,
                                         featF, featB,
                                         (l == NLAYER - 1) ? (float*)d_out : nullptr, N_NODES);
  }
}
